// FC_72232759984564
// MI455X (gfx1250) — compile-verified, matrix operand order checked by reading, not run on hardware
//
#include <hip/hip_runtime.h>
#include <stdint.h>

// ---------------------------------------------------------------------------
// Problem constants (from the reference)
// ---------------------------------------------------------------------------
constexpr int cNd = 50000, cNp = 20000, cEd = 65536, cEp = 65536, cB = 4096;
constexpr float cEPS = 1e-5f;

typedef __bf16 bf16;
typedef bf16  v16bf __attribute__((ext_vector_type(16)));
typedef bf16  v8bf  __attribute__((ext_vector_type(8)));
typedef float v8f   __attribute__((ext_vector_type(8)));

static __device__ __forceinline__ float leakyf(float x) {
    return x >= 0.f ? x : 0.01f * x;
}

// CDNA5 async copy: 16 bytes global -> LDS, tracked by ASYNCcnt (ISA 10.x / 08_async_tensor)
static __device__ __forceinline__ void async_copy16(unsigned lds_addr, const bf16* gaddr) {
    asm volatile("global_load_async_to_lds_b128 %0, %1, off"
                 :: "v"(lds_addr), "v"(gaddr)
                 : "memory");
}

#define GSLOOP(i, total)                                                     \
    for (size_t i = (size_t)blockIdx.x * blockDim.x + threadIdx.x;           \
         i < (total); i += (size_t)gridDim.x * blockDim.x)

// ---------------------------------------------------------------------------
// Small utility kernels
// ---------------------------------------------------------------------------
__global__ void k_zero_f32(float* __restrict__ p, size_t n) {
    GSLOOP(i, n) p[i] = 0.f;
}

// deg[dst] += w  (edge_index is [2,E] row-major: [0..E)=src, [E..2E)=dst)
__global__ void k_deg_accum(const int* __restrict__ ei, const float* __restrict__ ew,
                            float* __restrict__ deg, int E) {
    GSLOOP(e, (size_t)E) atomicAdd(&deg[ei[E + e]], ew[e]);
}

// dis[i] = rsqrt(deg[i] + 1)   (in place)
__global__ void k_deg_to_dis(float* __restrict__ d, int n) {
    GSLOOP(i, (size_t)n) d[i] = rsqrtf(d[i] + 1.0f);
}

// fp32 [rows x cols] -> bf16 [rows x colsp], zero-padded K
__global__ void k_cvt_bf16_pad(const float* __restrict__ src, bf16* __restrict__ dst,
                               int rows, int cols, int colsp) {
    size_t total = (size_t)rows * colsp;
    GSLOOP(i, total) {
        int c = (int)(i % colsp);
        int r = (int)(i / colsp);
        float v = (c < cols) ? src[(size_t)r * cols + c] : 0.f;
        dst[i] = (bf16)v;
    }
}

// W [K x N] fp32 -> Wt [N x Kp] bf16 (transposed, zero-padded K)
__global__ void k_cvt_bf16_t_pad(const float* __restrict__ W, bf16* __restrict__ Wt,
                                 int K, int N, int Kp) {
    size_t total = (size_t)N * Kp;
    GSLOOP(i, total) {
        int k = (int)(i % Kp);
        int n = (int)(i / Kp);
        float v = (k < K) ? W[(size_t)k * N + n] : 0.f;
        Wt[i] = (bf16)v;
    }
}

// ---------------------------------------------------------------------------
// bf16 WMMA GEMM with double-buffered async-LDS staging:
//   C[M x N] = A[M x Kp] * Bt[N x Kp]^T  (+ bias[n])
//   256 threads = 8 waves; wave tile 32x32 (2x2 of 16x16); block tile 64x128.
//   Per K-step (32): block stages A-tile 64x32 (4KB) + B-tile 128x32 (8KB)
//   into LDS with global_load_async_to_lds_b128 (3 ops/thread), double
//   buffered, synchronized with s_wait_asynccnt (async loads complete
//   in order, so "<=3" means the current tile is resident while the next
//   tile's 3 ops stay in flight).
//   16-bit fragment layout (ISA 7.12.2): lane l -> row (l&15),
//   kbase=(l>>4)*8, elems K = kbase + e + (e>=8 ? 8 : 0)
//   => two contiguous 16B ds loads at kbase and kbase+16.
// ---------------------------------------------------------------------------
__launch_bounds__(256)
__global__ void k_gemm_bf16_wmma(const bf16* __restrict__ A,
                                 const bf16* __restrict__ Bt,
                                 float* __restrict__ C,
                                 const float* __restrict__ bias,
                                 int M, int N, int Kp) {
    __shared__ bf16 As[2][64][32];    // 8 KB
    __shared__ bf16 Bs[2][128][32];   // 16 KB

    const int t    = threadIdx.x;
    const int lane = t & 31;
    const int wid  = t >> 5;
    const int m0b  = blockIdx.y * 64;
    const int n0b  = blockIdx.x * 128;

    // ---- staging assignments (one 16B chunk of A, two of B per thread) ----
    const int arow = t >> 2;              // 0..63
    const int acol = (t & 3) * 8;         // bf16 element offset, 16B chunks
    int ag = m0b + arow; if (ag > M - 1) ag = M - 1;     // clamp M edge
    const bf16* gA = A + (size_t)ag * Kp + acol;

    const int brow0 = t >> 2;             // 0..63
    const int brow1 = (t + 256) >> 2;     // 64..127
    const int bcol  = (t & 3) * 8;
    int bg0 = n0b + brow0; if (bg0 > N - 1) bg0 = N - 1;
    int bg1 = n0b + brow1; if (bg1 > N - 1) bg1 = N - 1;
    const bf16* gB0 = Bt + (size_t)bg0 * Kp + bcol;
    const bf16* gB1 = Bt + (size_t)bg1 * Kp + bcol;

    const int NT = Kp >> 5;

    // prefetch tile 0 into buffer 0
    async_copy16((unsigned)(uintptr_t)&As[0][arow][acol],  gA);
    async_copy16((unsigned)(uintptr_t)&Bs[0][brow0][bcol], gB0);
    async_copy16((unsigned)(uintptr_t)&Bs[0][brow1][bcol], gB1);

    // ---- compute-side fragment coordinates ----
    const int rm = (wid & 1) * 32;        // wave row offset within block tile
    const int cn = (wid >> 1) * 32;       // wave col offset within block tile
    const int lr = lane & 15;
    const int kb = (lane >> 4) * 8;

    v8f acc00 = {}, acc01 = {}, acc10 = {}, acc11 = {};

    for (int kt = 0; kt < NT; ++kt) {
        const int buf = kt & 1;
        if (kt + 1 < NT) {
            const int nxt = buf ^ 1;
            const int ke  = (kt + 1) << 5;
            async_copy16((unsigned)(uintptr_t)&As[nxt][arow][acol],  gA + ke);
            async_copy16((unsigned)(uintptr_t)&Bs[nxt][brow0][bcol], gB0 + ke);
            async_copy16((unsigned)(uintptr_t)&Bs[nxt][brow1][bcol], gB1 + ke);
            asm volatile("s_wait_asynccnt 0x3" ::: "memory");  // current tile resident
        } else {
            asm volatile("s_wait_asynccnt 0x0" ::: "memory");
        }
        __syncthreads();   // every wave's async ops for this tile are complete

        union { v16bf v; v8bf h[2]; } a0, a1, b0, b1;
        a0.h[0] = *(const v8bf*)&As[buf][rm + lr][kb];
        a0.h[1] = *(const v8bf*)&As[buf][rm + lr][kb + 16];
        a1.h[0] = *(const v8bf*)&As[buf][rm + 16 + lr][kb];
        a1.h[1] = *(const v8bf*)&As[buf][rm + 16 + lr][kb + 16];
        b0.h[0] = *(const v8bf*)&Bs[buf][cn + lr][kb];
        b0.h[1] = *(const v8bf*)&Bs[buf][cn + lr][kb + 16];
        b1.h[0] = *(const v8bf*)&Bs[buf][cn + 16 + lr][kb];
        b1.h[1] = *(const v8bf*)&Bs[buf][cn + 16 + lr][kb + 16];

        acc00 = __builtin_amdgcn_wmma_f32_16x16x32_bf16(false, a0.v, false, b0.v,
                                                        (short)0, acc00, false, false);
        acc01 = __builtin_amdgcn_wmma_f32_16x16x32_bf16(false, a0.v, false, b1.v,
                                                        (short)0, acc01, false, false);
        acc10 = __builtin_amdgcn_wmma_f32_16x16x32_bf16(false, a1.v, false, b0.v,
                                                        (short)0, acc10, false, false);
        acc11 = __builtin_amdgcn_wmma_f32_16x16x32_bf16(false, a1.v, false, b1.v,
                                                        (short)0, acc11, false, false);

        __syncthreads();   // all waves done reading `buf` before it is refilled
    }

    // ---- epilogue: bias + store ----
    const int m0  = m0b + rm;
    const int n0  = n0b + cn;
    const int cn0 = n0 + lr;
    const int cn1 = n0 + 16 + lr;
    const float bia0 = (bias != nullptr && cn0 < N) ? bias[cn0] : 0.f;
    const float bia1 = (bias != nullptr && cn1 < N) ? bias[cn1] : 0.f;

    // 32-bit C/D layout: VGPR j, lanes 0-15 -> M=j, lanes 16-31 -> M=j+8
    const int mbase = m0 + (lane >> 4) * 8;
#pragma unroll
    for (int j = 0; j < 8; ++j) {
        int mr = mbase + j;
        if (mr < M) {
            if (cn0 < N) C[(size_t)mr * N + cn0] = acc00[j] + bia0;
            if (cn1 < N) C[(size_t)mr * N + cn1] = acc01[j] + bia1;
        }
        int mr1 = mr + 16;
        if (mr1 < M) {
            if (cn0 < N) C[(size_t)mr1 * N + cn0] = acc10[j] + bia0;
            if (cn1 < N) C[(size_t)mr1 * N + cn1] = acc11[j] + bia1;
        }
    }
}

// ---------------------------------------------------------------------------
// GCN epilogue kernels
// ---------------------------------------------------------------------------
// agg[i][f] = dis[i]^2 * h[i][f] + bias[f]   (self-loop term + bias init)
__global__ void k_gcn_selfloop(const float* __restrict__ h, const float* __restrict__ dis,
                               const float* __restrict__ bias, float* __restrict__ agg,
                               int Nn, int F) {
    size_t total = (size_t)Nn * F;
    GSLOOP(i, total) {
        int f = (int)(i % F);
        int r = (int)(i / F);
        float d = dis[r];
        agg[i] = d * d * h[i] + bias[f];
    }
}

// one block per edge: agg[dst][f] += dis[src]*w*dis[dst] * h[src][f]
__global__ void k_gcn_scatter(const int* __restrict__ ei, const float* __restrict__ ew,
                              const float* __restrict__ dis, const float* __restrict__ h,
                              float* __restrict__ agg, int E, int F) {
    int e = blockIdx.x;
    int s = ei[e];
    int d = ei[E + e];
    float norm = dis[s] * ew[e] * dis[d];
    const float* hs = h + (size_t)s * F;
    float* ad = agg + (size_t)d * F;
    for (int f = threadIdx.x; f < F; f += blockDim.x)
        atomicAdd(&ad[f], norm * hs[f]);
}

// feature = concat(d_vecs, p_emb, leaky(aggd[d_index]), leaky(aggp[p_index])) -> bf16, K-padded
__global__ void k_build_feat(const float* __restrict__ dv, const float* __restrict__ pe,
                             const float* __restrict__ aggd, const float* __restrict__ aggp,
                             const int* __restrict__ di, const int* __restrict__ pi,
                             bf16* __restrict__ feat, int rows, int colsp) {
    size_t total = (size_t)rows * colsp;
    GSLOOP(i, total) {
        int c = (int)(i % colsp);
        int r = (int)(i / colsp);
        float v;
        if (c < 300)        v = dv[(size_t)r * 300 + c];
        else if (c < 1324)  v = pe[(size_t)r * 1024 + (c - 300)];
        else if (c < 2348)  v = leakyf(aggd[(size_t)di[r] * 1024 + (c - 1324)]);
        else if (c < 3372)  v = leakyf(aggp[(size_t)pi[r] * 1024 + (c - 2348)]);
        else                v = 0.f;
        feat[i] = (bf16)v;
    }
}

// ---------------------------------------------------------------------------
// BatchNorm (training-mode, biased variance) kernels
// ---------------------------------------------------------------------------
__global__ void k_bn_stats(const float* __restrict__ z, float* __restrict__ mean,
                           float* __restrict__ varr, int rows, int cols, int leaky_first) {
    __shared__ float s1[256], s2[256];
    int c = blockIdx.x;
    float a = 0.f, b = 0.f;
    for (int r = threadIdx.x; r < rows; r += 256) {
        float v = z[(size_t)r * cols + c];
        if (leaky_first) v = leakyf(v);
        a += v; b += v * v;
    }
    s1[threadIdx.x] = a; s2[threadIdx.x] = b;
    __syncthreads();
    for (int off = 128; off > 0; off >>= 1) {
        if ((int)threadIdx.x < off) {
            s1[threadIdx.x] += s1[threadIdx.x + off];
            s2[threadIdx.x] += s2[threadIdx.x + off];
        }
        __syncthreads();
    }
    if (threadIdx.x == 0) {
        float m = s1[0] / rows;
        mean[c] = m;
        varr[c] = s2[0] / rows - m * m;
    }
}

// y = leaky(BN(z)); write bf16 (next GEMM A) and optionally fp32 (feature output)
__global__ void k_bn_leaky_store(const float* __restrict__ z, const float* __restrict__ mean,
                                 const float* __restrict__ varr, const float* __restrict__ g,
                                 const float* __restrict__ be, bf16* __restrict__ outb,
                                 float* __restrict__ outf, int rows, int cols) {
    size_t total = (size_t)rows * cols;
    GSLOOP(i, total) {
        int c = (int)(i % cols);
        float y = (z[i] - mean[c]) * rsqrtf(varr[c] + cEPS) * g[c] + be[c];
        y = leakyf(y);
        outb[i] = (bf16)y;
        if (outf) outf[i] = y;
    }
}

// z = BN(leaky(z)) in place (stats were computed with leaky_first=1)
__global__ void k_leaky_bn_inplace(float* __restrict__ z, const float* __restrict__ mean,
                                   const float* __restrict__ varr, const float* __restrict__ g,
                                   const float* __restrict__ be, int rows, int cols) {
    size_t total = (size_t)rows * cols;
    GSLOOP(i, total) {
        int c = (int)(i % cols);
        float t = leakyf(z[i]);
        z[i] = (t - mean[c]) * rsqrtf(varr[c] + cEPS) * g[c] + be[c];
    }
}

// y[r] = dot(o[r,:], w) + b0   (one block per row, cols <= 256)
__global__ void k_fc_out(const float* __restrict__ o, const float* __restrict__ w,
                         const float* __restrict__ b0, float* __restrict__ y, int cols) {
    __shared__ float sh[256];
    int r = blockIdx.x;
    float v = ((int)threadIdx.x < cols) ? o[(size_t)r * cols + threadIdx.x] * w[threadIdx.x] : 0.f;
    sh[threadIdx.x] = v;
    __syncthreads();
    for (int off = 128; off > 0; off >>= 1) {
        if ((int)threadIdx.x < off) sh[threadIdx.x] += sh[threadIdx.x + off];
        __syncthreads();
    }
    if (threadIdx.x == 0) y[r] = sh[0] + b0[0];
}

// ---------------------------------------------------------------------------
// Host orchestration
// ---------------------------------------------------------------------------
extern "C" void kernel_launch(void* const* d_in, const int* in_sizes, int n_in,
                              void* d_out, int out_size, void* d_ws, size_t ws_size,
                              hipStream_t stream) {
    (void)in_sizes; (void)n_in; (void)out_size; (void)ws_size;

    const int*   d_index = (const int*)d_in[0];
    const int*   p_index = (const int*)d_in[1];
    const float* d_vecs  = (const float*)d_in[2];
    const float* p_emb   = (const float*)d_in[3];
    const float* d_ecfps = (const float*)d_in[4];
    const int*   d_ei    = (const int*)d_in[5];
    const float* d_ew    = (const float*)d_in[6];
    const float* p_gos   = (const float*)d_in[7];
    const int*   p_ei    = (const int*)d_in[8];
    const float* p_ew    = (const float*)d_in[9];
    const float* W_dg  = (const float*)d_in[10]; const float* b_dg  = (const float*)d_in[11];
    const float* W_pg  = (const float*)d_in[12]; const float* b_pg  = (const float*)d_in[13];
    const float* W_e1  = (const float*)d_in[14]; const float* b_e1  = (const float*)d_in[15];
    const float* g_e1  = (const float*)d_in[16]; const float* be_e1 = (const float*)d_in[17];
    const float* W_e2  = (const float*)d_in[18]; const float* b_e2  = (const float*)d_in[19];
    const float* g_e2  = (const float*)d_in[20]; const float* be_e2 = (const float*)d_in[21];
    const float* W_o1  = (const float*)d_in[22]; const float* b_o1  = (const float*)d_in[23];
    const float* g_o   = (const float*)d_in[24]; const float* be_o  = (const float*)d_in[25];
    const float* W_o2  = (const float*)d_in[26]; const float* b_o2  = (const float*)d_in[27];

    float* y_out    = (float*)d_out;          // [4096]
    float* feat_out = (float*)d_out + cB;     // [4096 x 1024]

    // ---- workspace bump allocator (256B aligned regions) ----
    char* wsp = (char*)d_ws;
    auto take = [&](size_t bytes) -> void* {
        void* r = (void*)wsp;
        wsp += (bytes + 255) & ~(size_t)255;
        return r;
    };
    float* dis_d = (float*)take((size_t)cNd * 4);
    float* dis_p = (float*)take((size_t)cNp * 4);
    float* mean  = (float*)take(2048 * 4);
    float* varr  = (float*)take(2048 * 4);
    bf16*  Abuf  = (bf16*) take((size_t)cNp * 2816 * 2);   // fits A_d (50000x1024) too
    bf16*  Wt    = (bf16*) take((size_t)3392 * 2048 * 2);  // largest transposed weight
    float* Hd    = (float*)take((size_t)cNd * 1024 * 4);   // drug h; protein h+agg overlay
    float* Aggd  = (float*)take((size_t)cNd * 1024 * 4);
    bf16*  featb = (bf16*) take((size_t)cB * 3392 * 2);
    float* Z     = (float*)take((size_t)cB * 2048 * 4);    // z1 / z2 / z3 (sequential reuse)
    bf16*  Act   = (bf16*) take((size_t)cB * 2048 * 2);    // a1 / a2 (sequential reuse)
    float* Hp    = Hd;                                      // 20000x1024
    float* Aggp  = Hd + (size_t)cNp * 1024;                 // 20000x1024 (fits in Hd region)

    dim3 b256(256);
    auto nb = [](size_t total) -> unsigned {
        size_t n = (total + 255) / 256;
        if (n > (size_t)1048576) n = 1048576;
        return (unsigned)n;
    };

    // ================= Drug GCN branch =================
    k_zero_f32     <<<nb(cNd), b256, 0, stream>>>(dis_d, cNd);
    k_deg_accum    <<<nb(cEd), b256, 0, stream>>>(d_ei, d_ew, dis_d, cEd);
    k_deg_to_dis   <<<nb(cNd), b256, 0, stream>>>(dis_d, cNd);
    k_cvt_bf16_pad <<<nb((size_t)cNd * 1024), b256, 0, stream>>>(d_ecfps, Abuf, cNd, 1024, 1024);
    k_cvt_bf16_t_pad<<<nb((size_t)1024 * 1024), b256, 0, stream>>>(W_dg, Wt, 1024, 1024, 1024);
    {
        dim3 g(1024 / 128, (cNd + 63) / 64);
        k_gemm_bf16_wmma<<<g, b256, 0, stream>>>(Abuf, Wt, Hd, nullptr, cNd, 1024, 1024);
    }
    k_gcn_selfloop <<<nb((size_t)cNd * 1024), b256, 0, stream>>>(Hd, dis_d, b_dg, Aggd, cNd, 1024);
    k_gcn_scatter  <<<dim3(cEd), b256, 0, stream>>>(d_ei, d_ew, dis_d, Hd, Aggd, cEd, 1024);

    // ================= Protein GCN branch =================
    k_zero_f32     <<<nb(cNp), b256, 0, stream>>>(dis_p, cNp);
    k_deg_accum    <<<nb(cEp), b256, 0, stream>>>(p_ei, p_ew, dis_p, cEp);
    k_deg_to_dis   <<<nb(cNp), b256, 0, stream>>>(dis_p, cNp);
    k_cvt_bf16_pad <<<nb((size_t)cNp * 2816), b256, 0, stream>>>(p_gos, Abuf, cNp, 2812, 2816);
    k_cvt_bf16_t_pad<<<nb((size_t)1024 * 2816), b256, 0, stream>>>(W_pg, Wt, 2812, 1024, 2816);
    {
        dim3 g(1024 / 128, (cNp + 63) / 64);
        k_gemm_bf16_wmma<<<g, b256, 0, stream>>>(Abuf, Wt, Hp, nullptr, cNp, 1024, 2816);
    }
    k_gcn_selfloop <<<nb((size_t)cNp * 1024), b256, 0, stream>>>(Hp, dis_p, b_pg, Aggp, cNp, 1024);
    k_gcn_scatter  <<<dim3(cEp), b256, 0, stream>>>(p_ei, p_ew, dis_p, Hp, Aggp, cEp, 1024);

    // ================= Concatenated feature =================
    k_build_feat   <<<nb((size_t)cB * 3392), b256, 0, stream>>>(
        d_vecs, p_emb, Aggd, Aggp, d_index, p_index, featb, cB, 3392);

    // ================= FC1 (3372->2048) + BN + leaky =================
    k_cvt_bf16_t_pad<<<nb((size_t)2048 * 3392), b256, 0, stream>>>(W_e1, Wt, 3372, 2048, 3392);
    {
        dim3 g(2048 / 128, cB / 64);
        k_gemm_bf16_wmma<<<g, b256, 0, stream>>>(featb, Wt, Z, b_e1, cB, 2048, 3392);
    }
    k_bn_stats     <<<dim3(2048), b256, 0, stream>>>(Z, mean, varr, cB, 2048, 0);
    k_bn_leaky_store<<<nb((size_t)cB * 2048), b256, 0, stream>>>(
        Z, mean, varr, g_e1, be_e1, Act, nullptr, cB, 2048);

    // ================= FC2 (2048->1024) + BN + leaky -> feature =================
    k_cvt_bf16_t_pad<<<nb((size_t)1024 * 2048), b256, 0, stream>>>(W_e2, Wt, 2048, 1024, 2048);
    {
        dim3 g(1024 / 128, cB / 64);
        k_gemm_bf16_wmma<<<g, b256, 0, stream>>>(Act, Wt, Z, b_e2, cB, 1024, 2048);
    }
    k_bn_stats     <<<dim3(1024), b256, 0, stream>>>(Z, mean, varr, cB, 1024, 0);
    k_bn_leaky_store<<<nb((size_t)cB * 1024), b256, 0, stream>>>(
        Z, mean, varr, g_e2, be_e2, Act, feat_out, cB, 1024);

    // ================= FC3 (1024->256), leaky then BN =================
    k_cvt_bf16_t_pad<<<nb((size_t)256 * 1024), b256, 0, stream>>>(W_o1, Wt, 1024, 256, 1024);
    {
        dim3 g(256 / 128, cB / 64);
        k_gemm_bf16_wmma<<<g, b256, 0, stream>>>(Act, Wt, Z, b_o1, cB, 256, 1024);
    }
    k_bn_stats     <<<dim3(256), b256, 0, stream>>>(Z, mean, varr, cB, 256, 1);
    k_leaky_bn_inplace<<<nb((size_t)cB * 256), b256, 0, stream>>>(Z, mean, varr, g_o, be_o, cB, 256);

    // ================= FC4 (256->1) =================
    k_fc_out       <<<dim3(cB), b256, 0, stream>>>(Z, W_o2, b_o2, y_out, 256);
}
